// ParaConv2d_47347719471159
// MI455X (gfx1250) — compile-verified, matrix operand order checked by reading, not run on hardware
//
#include <hip/hip_runtime.h>
#include <hip/hip_bf16.h>
#include <math.h>

// ---------------------------------------------------------------------------
// ParaConv2d on gfx1250: implicit-GEMM with V_WMMA_F32_16X16X32_F16.
//   GEMM: [pixels x 576] x [576 x 256]  (cols 0..127 = h conv, 128..255 = p conv)
//   Epilogue: out = 0.1*(hx + h0 - sqrt(max(xsq - 2*px + (|p|^2+p0^2), 1e-12)))
// Each wave: 32 pixels x 128 cols (2 row tiles x 8 col tiles) so every B
// fragment loaded from cache feeds TWO wmma ops (register-level B reuse).
// ---------------------------------------------------------------------------

typedef __attribute__((ext_vector_type(16))) _Float16 v16h;
typedef __attribute__((ext_vector_type(8)))  float    v8f;
typedef __attribute__((ext_vector_type(4)))  float    v4f;

union Frag16 { v16h h; v4f q[2]; };

#define C_IN   64
#define M_OUT  128
#define DDIM   576          // 64*9, = 18 * 32
#define HW     112
#define HWP    114
#define PIX    (HW*HW)      // 12544
#define NIMG   32
#define ROWS_B 128          // pixels per block (grid: 12544/128 = 98)
#define KC     192          // K chunk staged in LDS (6 wmma k-steps, 3 chunks)
#define LDA    200          // f16 stride for LDS A (192 + 8 pad; 400B rows, 16B aligned)

// ---- workspace layout (bytes) ----
#define WS_XPH   0                                   // f16  32*64*114*114  = 53,231,616 B
#define WS_S     (53231616)                          // f32  32*114*114     =  1,663,488 B
#define WS_WFRAG (WS_S + 1663488)                    // f16  18*16*512      =    294,912 B
#define WS_CVEC  (WS_WFRAG + 294912)                 // f32  256            =      1,024 B

// ---------------------------------------------------------------------------
// prep_x: xph = f16(0.01*x) zero-padded; s[n][yy][xx] = sum_c xph^2
// grid (114, 32), block 128
// ---------------------------------------------------------------------------
__global__ void prep_x_kernel(const float* __restrict__ x,
                              _Float16* __restrict__ xph,
                              float* __restrict__ s) {
    const int xx = threadIdx.x;
    const int yy = blockIdx.x;
    const int n  = blockIdx.y;
    if (xx >= HWP) return;
    const bool interior = (yy >= 1) && (yy <= HW) && (xx >= 1) && (xx <= HW);
    const float* xb = x + (size_t)n * C_IN * PIX;
    _Float16* ob = xph + ((size_t)n * C_IN * HWP + yy) * HWP + xx;
    const int xoff = (yy - 1) * HW + (xx - 1);
    float acc = 0.0f;
    #pragma unroll 4
    for (int c = 0; c < C_IN; ++c) {
        float v = interior ? 0.01f * xb[(size_t)c * PIX + xoff] : 0.0f;
        ob[(size_t)c * HWP * HWP] = (_Float16)v;
        acc += v * v;
    }
    s[((size_t)n * HWP + yy) * HWP + xx] = acc;
}

// ---------------------------------------------------------------------------
// prep_w: concatenated [576 x 256] weights, stored in B-fragment order:
//   half index = (((ks*16 + ct)*32 + lane)*2 + part)*8 + e
//   col = ct*16 + lane%16,  k = ks*32 + part*16 + (lane>=16 ? 8 : 0) + e
// grid 576 blocks x 256 threads (18*16*512 = 147456 elements)
// ---------------------------------------------------------------------------
__global__ void prep_w_kernel(const float* __restrict__ pars,
                              _Float16* __restrict__ wfrag) {
    const int tid  = blockIdx.x * 256 + threadIdx.x;
    if (tid >= 18 * 16 * 512) return;
    const int e    = tid & 7;
    const int part = (tid >> 3) & 1;
    const int lane = (tid >> 4) & 31;
    const int ct   = (tid >> 9) & 15;
    const int ks   = tid >> 13;
    const int k    = ks * 32 + part * 16 + ((lane >> 4) & 1) * 8 + e;
    const int col  = ct * 16 + (lane & 15);
    float v;
    if (col < M_OUT) v = pars[M_OUT + col * DDIM + k];                          // h weights
    else             v = pars[M_OUT + M_OUT * DDIM + (col - M_OUT) * DDIM + k]; // p weights
    wfrag[tid] = (_Float16)v;
}

// ---------------------------------------------------------------------------
// prep_c: cvec[0:128] = h0 ; cvec[128:256] = |p_m|^2 + p0_m^2
// grid 256 blocks x 64 threads
// ---------------------------------------------------------------------------
__global__ void prep_c_kernel(const float* __restrict__ pars,
                              float* __restrict__ cvec) {
    __shared__ float red[64];
    const int j = blockIdx.x;
    const int tid = threadIdx.x;
    if (j < M_OUT) {
        if (tid == 0) cvec[j] = pars[j];
        return;
    }
    const int jj = j - M_OUT;
    const float* pw = pars + M_OUT + M_OUT * DDIM + jj * DDIM;
    float a = 0.0f;
    for (int k = tid; k < DDIM; k += 64) { float v = pw[k]; a += v * v; }
    red[tid] = a;
    __syncthreads();
    for (int st = 32; st > 0; st >>= 1) {
        if (tid < st) red[tid] += red[tid + st];
        __syncthreads();
    }
    if (tid == 0) {
        float p0 = pars[M_OUT + 2 * M_OUT * DDIM + jj];
        cvec[j] = red[0] + p0 * p0;
    }
}

// ---------------------------------------------------------------------------
// main: block = 256 threads (8 waves), tile = 128 pixels x 256 cols.
// wave w: row group rg = w>>1 (32 pixels = 2 row tiles), col half pcol = w&1.
// acc[ra*8 + t]     (t=0..3): h cols 64*pcol + 16t, row tile ra
// acc[ra*8 + 4 + t] (t=0..3): p cols 64*pcol + 16t, row tile ra
// ---------------------------------------------------------------------------
__global__ __launch_bounds__(256) void paraconv_gemm_kernel(
        const _Float16* __restrict__ xph,
        const float*    __restrict__ s,
        const _Float16* __restrict__ wfrag,
        const float*    __restrict__ cvec,
        float*          __restrict__ out) {
    extern __shared__ char smem[];
    _Float16* ldsA  = (_Float16*)smem;                        // 128 * LDA f16 = 51,200 B
    float*    ldsS9 = (float*)(smem + ROWS_B * LDA * 2);      // 128 f32

    const int n       = blockIdx.y;
    const int pixBase = blockIdx.x * ROWS_B;
    const int tid     = threadIdx.x;
    const int lane    = tid & 31;
    const int w       = tid >> 5;
    const int rg      = w >> 1;       // 0..3, 32 pixels each
    const int pcol    = w & 1;

    const _Float16* xin = xph + (size_t)n * C_IN * HWP * HWP;

    // --- stage 3x3 window sums of per-pixel |x|^2 (xsq term) ---
    if (tid < ROWS_B) {
        const int p = pixBase + tid;
        const int y = p / HW;
        const int xc = p - y * HW;
        const float* sn = s + (size_t)n * HWP * HWP;
        float a = 0.0f;
        #pragma unroll
        for (int ky = 0; ky < 3; ++ky)
            #pragma unroll
            for (int kx = 0; kx < 3; ++kx)
                a += sn[(size_t)(y + ky) * HWP + (xc + kx)];
        ldsS9[tid] = a;
    }

    v8f acc[16] = {};

    const int rowBaseL = rg * 32 + (lane & 15);
    const int kHiOff   = ((lane >> 4) & 1) * 8;

    for (int chunk = 0; chunk < 3; ++chunk) {
        __syncthreads();
        // --- im2col stage of A chunk: 128 rows x 192 K (f16) ---
        #pragma unroll 4
        for (int it = 0; it < (ROWS_B * KC) / 256; ++it) {   // 96 iters
            const int idx = it * 256 + tid;
            const int row = idx & (ROWS_B - 1);
            const int kl  = idx >> 7;                // 0..191
            const int k   = chunk * KC + kl;
            const int p   = pixBase + row;
            const int y   = p / HW;
            const int xc  = p - y * HW;
            const int c   = k / 9;
            const int r9  = k - c * 9;
            const int ky  = r9 / 3;
            const int kx  = r9 - ky * 3;
            ldsA[row * LDA + kl] =
                xin[((size_t)c * HWP + (y + ky)) * HWP + (xc + kx)];
        }
        __syncthreads();

        // --- 6 wmma k-steps over this chunk ---
        for (int ksl = 0; ksl < 6; ++ksl) {
            Frag16 a0, a1;
            const _Float16* ap = ldsA + rowBaseL * LDA + ksl * 32 + kHiOff;
            a0.q[0] = *(const v4f*)(ap);
            a0.q[1] = *(const v4f*)(ap + 16);
            a1.q[0] = *(const v4f*)(ap + 16 * LDA);
            a1.q[1] = *(const v4f*)(ap + 16 * LDA + 16);

            const int ks = chunk * 6 + ksl;
            #pragma unroll
            for (int t = 0; t < 8; ++t) {
                const int ct = pcol * 4 + ((t < 4) ? t : (t + 4));
                Frag16 b;
                const v4f* wp = (const v4f*)wfrag +
                                (((size_t)ks * 16 + ct) * 32 + lane) * 2;
                b.q[0] = wp[0];
                b.q[1] = wp[1];
                acc[t] = __builtin_amdgcn_wmma_f32_16x16x32_f16(
                    false, a0.h, false, b.h, (short)0, acc[t], false, false);
                acc[8 + t] = __builtin_amdgcn_wmma_f32_16x16x32_f16(
                    false, a1.h, false, b.h, (short)0, acc[8 + t], false, false);
            }
        }
    }

    // --- fused epilogue: out = 0.1*(hx + h0 - sqrt(max(xsq - 2*px + cP, eps))) ---
    const int mloc = lane & 15;
    const int hi   = (lane >> 4) & 1;
    #pragma unroll
    for (int ra = 0; ra < 2; ++ra) {
        #pragma unroll
        for (int t = 0; t < 4; ++t) {
            const int m   = pcol * 64 + t * 16 + mloc;
            const float cvH = cvec[m];
            const float cvP = cvec[M_OUT + m];
            float o[8];
            #pragma unroll
            for (int i = 0; i < 8; ++i) {
                const float s9 = ldsS9[rg * 32 + ra * 16 + hi * 8 + i];
                const float hx = acc[ra * 8 + t][i] + cvH;
                const float d2 = s9 - 2.0f * acc[ra * 8 + 4 + t][i] + cvP;
                const float d  = sqrtf(fmaxf(d2, 1e-12f));
                o[i] = 0.1f * (hx - d);
            }
            const int p0 = pixBase + rg * 32 + ra * 16 + hi * 8;
            float* op = out + ((size_t)(n * M_OUT + m)) * PIX + p0;
            v4f v0 = { o[0], o[1], o[2], o[3] };
            v4f v1 = { o[4], o[5], o[6], o[7] };
            *(v4f*)(op)     = v0;
            *(v4f*)(op + 4) = v1;
        }
    }
}

// ---------------------------------------------------------------------------
extern "C" void kernel_launch(void* const* d_in, const int* in_sizes, int n_in,
                              void* d_out, int out_size, void* d_ws, size_t ws_size,
                              hipStream_t stream) {
    const float* x    = (const float*)d_in[0];
    const float* pars = (const float*)d_in[1];
    float* out = (float*)d_out;
    char* ws = (char*)d_ws;

    _Float16* xph   = (_Float16*)(ws + WS_XPH);
    float*    s     = (float*)(ws + WS_S);
    _Float16* wfrag = (_Float16*)(ws + WS_WFRAG);
    float*    cvec  = (float*)(ws + WS_CVEC);

    prep_x_kernel<<<dim3(HWP, NIMG), 128, 0, stream>>>(x, xph, s);
    prep_w_kernel<<<576, 256, 0, stream>>>(pars, wfrag);
    prep_c_kernel<<<256, 64, 0, stream>>>(pars, cvec);

    const size_t shmem = (size_t)ROWS_B * LDA * 2 + ROWS_B * sizeof(float); // 51,712 B
    paraconv_gemm_kernel<<<dim3(PIX / ROWS_B, NIMG), 256, shmem, stream>>>(
        xph, s, wfrag, cvec, out);
}
